// Informer_20186346291963
// MI455X (gfx1250) — matrix operand order repaired and checked by reading, not compiled, not run on hardware
//
#include <hip/hip_runtime.h>

// ---------------------------------------------------------------------------
// Informer forward for MI455X (gfx1250, wave32, WMMA).
//  * U == N == 15 -> ProbSparse == softmax(softmax(Q K^T / 8))
//  * memory_proj output unused -> skipped
//  * Batch elements independent -> chunk whole pipeline over batch (ws-sized)
// GEMMs: bf16 operands via v_wmma_f32_16x16x32_bf16, f32 accumulate.
// Producers write bf16 activation mirrors so the GEMM hot loop does NO
// format conversion; LDS staging uses global_load_async_to_lds_b128 +
// s_wait_asynccnt (double buffered) when the toolchain exposes the builtins.
// ---------------------------------------------------------------------------

typedef __attribute__((ext_vector_type(16))) __bf16 v16bf;
typedef __attribute__((ext_vector_type(8)))  float  v8f;
typedef __attribute__((ext_vector_type(4)))  int    v4i;

struct __align__(16) U4 { unsigned int x, y, z, w; };

union Frag16 { v16bf v; U4 q[2]; };

__device__ __forceinline__ unsigned int f2bf(float f) {
  unsigned int u = __float_as_uint(f);
  u += 0x7FFFu + ((u >> 16) & 1u);   // round-to-nearest-even
  return u >> 16;
}
__device__ __forceinline__ float bf2f(unsigned short u) {
  return __uint_as_float(((unsigned int)u) << 16);
}

// ------------------- CDNA5 async global->LDS copy (guarded) -----------------
#if __has_builtin(__builtin_amdgcn_global_load_async_to_lds_b128) && \
    __has_builtin(__builtin_amdgcn_s_wait_asynccnt)
#define HAS_ASYNC_LDS 1
#else
#define HAS_ASYNC_LDS 0
#endif

typedef __attribute__((address_space(1))) v4i as1_v4i;  // global int4
typedef __attribute__((address_space(3))) v4i as3_v4i;  // LDS int4

__device__ __forceinline__ void cp16_g2l(const unsigned short* g,
                                         unsigned short* l) {
#if HAS_ASYNC_LDS
  __builtin_amdgcn_global_load_async_to_lds_b128(
      (as1_v4i*)(size_t)g, (as3_v4i*)(unsigned)(size_t)l, 0, 0);
#else
  *(U4*)l = *(const U4*)g;
#endif
}
__device__ __forceinline__ void cp_wait() {
#if HAS_ASYNC_LDS
  __builtin_amdgcn_s_wait_asynccnt(0);
#endif
}

// ------------------------- weight convert + transpose -----------------------
// w: f32 [K][N] row-major  ->  wt: bf16 bits [N][K] row-major
__global__ void k_wconv_t(const float* __restrict__ w,
                          unsigned short* __restrict__ wt, int K, int N) {
  int idx = blockIdx.x * 256 + threadIdx.x;
  if (idx >= K * N) return;
  int k = idx / N, n = idx % N;
  wt[(size_t)n * K + k] = (unsigned short)f2bf(w[idx]);
}

// ------------------------------ input projection ----------------------------
__global__ void __launch_bounds__(128) k_input_proj(
    const float* __restrict__ x, const float* __restrict__ W,
    const float* __restrict__ b, const float* __restrict__ pe,
    float* __restrict__ h, unsigned short* __restrict__ hbf) {
  __shared__ float xs[16];
  int t = blockIdx.x;
  if (threadIdx.x < 16) xs[threadIdx.x] = x[(size_t)t * 16 + threadIdx.x];
  __syncthreads();
  int pr = t % 15;
#pragma unroll
  for (int j = 0; j < 4; ++j) {
    int d = threadIdx.x * 4 + j;
    float acc = b[d] + pe[pr * 512 + d];
#pragma unroll
    for (int k = 0; k < 16; ++k) acc += xs[k] * W[k * 512 + d];
    h[(size_t)t * 512 + d] = acc;
    hbf[(size_t)t * 512 + d] = (unsigned short)f2bf(acc);
  }
}

// --------------------------------- bf16 GEMM --------------------------------
// C[M,N] = act( A[M,K](bf16) @ Wt[N,K](bf16)^T + bias[N] )
// mode bit0 = relu, bit1 = bf16 output (else f32)
#define BM 64
#define BN 64
#define A_STR 40   // ushorts; multiple of 8 -> 16B-aligned rows
#define B_STR 40

__global__ void __launch_bounds__(256) k_gemm_bf16(
    const unsigned short* __restrict__ A, const unsigned short* __restrict__ Wt,
    const float* __restrict__ bias, void* __restrict__ Cout,
    int M, int N, int K, int mode) {
  __shared__ __align__(16) unsigned short As[2][BM * A_STR];
  __shared__ __align__(16) unsigned short Bs[2][BN * B_STR];

  const int tid  = threadIdx.x;
  const int lane = tid & 31, wave = tid >> 5;
  const int m0 = blockIdx.x * BM, n0 = blockIdx.y * BN;
  const int wm = (wave & 3) * 16;
  const int wn = (wave >> 2) * 32;

  v8f c0 = {};
  v8f c1 = {};

  // cooperative tile-load coordinates: 64 rows x 32 cols bf16, 16B per thread
  const int lr = tid >> 2;               // 0..63
  const int lc = (tid & 3) * 8;          // 0,8,16,24
  const unsigned short* Aptr = A  + (size_t)(m0 + lr) * K + lc;  // padded rows
  const unsigned short* Bptr = Wt + (size_t)(n0 + lr) * K + lc;
  unsigned short* Adst0 = &As[0][lr * A_STR + lc];
  unsigned short* Adst1 = &As[1][lr * A_STR + lc];
  unsigned short* Bdst0 = &Bs[0][lr * B_STR + lc];
  unsigned short* Bdst1 = &Bs[1][lr * B_STR + lc];

  // fragment addresses (contiguous k = 16*(lane>>4) + s, same order for A and B)
  const int fr   = wm + (lane & 15);         // A row
  const int koff = (lane >> 4) * 16;         // A K sub-offset
  const int cn   = wn + (lane & 15);         // B column
  const int k0   = (lane >> 4) * 16;         // B K base

  const int ntiles = K >> 5;
  // preload tile 0 into buffer 0
  cp16_g2l(Aptr, Adst0);
  cp16_g2l(Bptr, Bdst0);
  cp_wait();
  __syncthreads();

  for (int kt = 0; kt < ntiles; ++kt) {
    const int cur = kt & 1;
    if (kt + 1 < ntiles) {               // prefetch next tile into other buffer
      unsigned short* Ad = cur ? Adst0 : Adst1;
      unsigned short* Bd = cur ? Bdst0 : Bdst1;
      cp16_g2l(Aptr + (kt + 1) * 32, Ad);
      cp16_g2l(Bptr + (kt + 1) * 32, Bd);
    }
    const unsigned short* Ab = As[cur];
    const unsigned short* Bb = Bs[cur];
    Frag16 af, b0f, b1f;
    af.q[0]  = *(const U4*)&Ab[fr * A_STR + koff];
    af.q[1]  = *(const U4*)&Ab[fr * A_STR + koff + 8];
    b0f.q[0] = *(const U4*)&Bb[cn * B_STR + k0];
    b0f.q[1] = *(const U4*)&Bb[cn * B_STR + k0 + 8];
    b1f.q[0] = *(const U4*)&Bb[(cn + 16) * B_STR + k0];
    b1f.q[1] = *(const U4*)&Bb[(cn + 16) * B_STR + k0 + 8];

    c0 = __builtin_amdgcn_wmma_f32_16x16x32_bf16(false, af.v, false, b0f.v,
                                                 (short)0, c0, false, false);
    c1 = __builtin_amdgcn_wmma_f32_16x16x32_bf16(false, af.v, false, b1f.v,
                                                 (short)0, c1, false, false);
    cp_wait();          // next-tile LDS writes complete (per wave)
    __syncthreads();    // visible to all waves; cur buffer free for reuse
  }

  // epilogue: lane = col (lane&15), VGPR r = row r + (lane>=16)*8
  const int colb = n0 + wn + (lane & 15);
  const int rowb = m0 + wm + ((lane >> 4) * 8);
  const float bias0 = bias[colb], bias1 = bias[colb + 16];
#pragma unroll
  for (int r = 0; r < 8; ++r) {
    int row = rowb + r;
    if (row < M) {
      float v0 = c0[r] + bias0;
      float v1 = c1[r] + bias1;
      if (mode & 1) { v0 = fmaxf(v0, 0.f); v1 = fmaxf(v1, 0.f); }
      if (mode & 2) {
        unsigned short* C = (unsigned short*)Cout;
        C[(size_t)row * N + colb]      = (unsigned short)f2bf(v0);
        C[(size_t)row * N + colb + 16] = (unsigned short)f2bf(v1);
      } else {
        float* C = (float*)Cout;
        C[(size_t)row * N + colb]      = v0;
        C[(size_t)row * N + colb + 16] = v1;
      }
    }
  }
}

// ------------------------------ attention core ------------------------------
// qkv: bf16 [Mc,1536] rows = [q(8x64)|k(8x64)|v(8x64)]; out: bf16 [Mc,512]
// per (b,h): S = qk^T/8; A = softmax(softmax(S)); out = A @ v
__global__ void __launch_bounds__(128) k_attn(
    const unsigned short* __restrict__ qkv, unsigned short* __restrict__ out,
    int Bc) {
  __shared__ float sh[4 * 3136];     // per wave: q960 | k960 | v960 | S256
  const int wave = threadIdx.x >> 5, lane = threadIdx.x & 31;
  const int unit = blockIdx.x * 4 + wave;
  const int b = unit >> 3, hh = unit & 7;
  const bool act = (b < Bc);
  float* q = &sh[wave * 3136];
  float* k = q + 960;
  float* v = k + 960;
  float* S = v + 960;

  if (act) {
    for (int i = lane; i < 960; i += 32) {
      int m = i >> 6, d = i & 63;
      size_t rb = (size_t)(b * 15 + m) * 1536;
      q[i] = bf2f(qkv[rb + hh * 64 + d]);
      k[i] = bf2f(qkv[rb + 512 + hh * 64 + d]);
      v[i] = bf2f(qkv[rb + 1024 + hh * 64 + d]);
    }
  }
  __syncthreads();
  if (act) {
    for (int p = lane; p < 225; p += 32) {
      int i = p / 15, j = p % 15;
      float s = 0.f;
#pragma unroll
      for (int d = 0; d < 64; ++d) s += q[i * 64 + d] * k[j * 64 + d];
      S[i * 16 + j] = s * 0.125f;    // Dh^-0.5, Dh=64
    }
  }
  __syncthreads();
#pragma unroll
  for (int pass = 0; pass < 2; ++pass) {   // double softmax (ProbSparse, U==N)
    if (act && lane < 15) {
      float mx = -1e30f;
#pragma unroll
      for (int j = 0; j < 15; ++j) mx = fmaxf(mx, S[lane * 16 + j]);
      float e[15], sum = 0.f;
#pragma unroll
      for (int j = 0; j < 15; ++j) { e[j] = __expf(S[lane * 16 + j] - mx); sum += e[j]; }
      float inv = 1.f / sum;
#pragma unroll
      for (int j = 0; j < 15; ++j) S[lane * 16 + j] = e[j] * inv;
    }
    __syncthreads();
  }
  if (act) {
    for (int p = lane; p < 960; p += 32) {
      int m = p >> 6, d = p & 63;
      float s = 0.f;
#pragma unroll
      for (int j = 0; j < 15; ++j) s += S[m * 16 + j] * v[j * 64 + d];
      out[(size_t)(b * 15 + m) * 512 + hh * 64 + d] = (unsigned short)f2bf(s);
    }
  }
}

// --------------------------- residual + LayerNorm ---------------------------
// h = LN(h + t)*g + b ; also writes bf16 mirror. One wave per token.
__global__ void __launch_bounds__(256) k_add_ln(
    float* __restrict__ h, unsigned short* __restrict__ hbf,
    const float* __restrict__ t, const float* __restrict__ g,
    const float* __restrict__ be, int Mc) {
  const int wave = threadIdx.x >> 5, lane = threadIdx.x & 31;
  const int tok = blockIdx.x * 8 + wave;
  if (tok >= Mc) return;
  float* hp = h + (size_t)tok * 512;
  const float* tp = t + (size_t)tok * 512;
  float vals[16], s = 0.f;
#pragma unroll
  for (int i = 0; i < 16; ++i) {
    int d = lane + i * 32;
    float x = hp[d] + tp[d];
    vals[i] = x; s += x;
  }
#pragma unroll
  for (int off = 16; off >= 1; off >>= 1) s += __shfl_xor(s, off, 32);
  float mean = s * (1.f / 512.f), var = 0.f;
#pragma unroll
  for (int i = 0; i < 16; ++i) { float dd = vals[i] - mean; var += dd * dd; }
#pragma unroll
  for (int off = 16; off >= 1; off >>= 1) var += __shfl_xor(var, off, 32);
  float inv = rsqrtf(var * (1.f / 512.f) + 1e-5f);
  unsigned short* hb = hbf + (size_t)tok * 512;
#pragma unroll
  for (int i = 0; i < 16; ++i) {
    int d = lane + i * 32;
    float o = (vals[i] - mean) * inv * g[d] + be[d];
    hp[d] = o;
    hb[d] = (unsigned short)f2bf(o);
  }
}

// --------------------------- mean over seq + out proj -----------------------
__global__ void __launch_bounds__(256) k_mean_out(
    const float* __restrict__ h, const float* __restrict__ w,
    const float* __restrict__ b, float* __restrict__ out, int Bc) {
  const int wave = threadIdx.x >> 5, lane = threadIdx.x & 31;
  const int bb = blockIdx.x * 8 + wave;
  if (bb >= Bc) return;
  float acc = 0.f;
  for (int d = lane; d < 512; d += 32) {
    float s = 0.f;
#pragma unroll
    for (int n = 0; n < 15; ++n) s += h[(size_t)(bb * 15 + n) * 512 + d];
    acc += (s * (1.f / 15.f)) * w[d];
  }
#pragma unroll
  for (int off = 16; off >= 1; off >>= 1) acc += __shfl_xor(acc, off, 32);
  if (lane == 0) out[bb] = acc + b[0];
}

// =============================== host driver ================================

static inline int imin(int a, int b) { return a < b ? a : b; }

extern "C" void kernel_launch(void* const* d_in, const int* in_sizes, int n_in,
                              void* d_out, int out_size, void* d_ws, size_t ws_size,
                              hipStream_t stream) {
  (void)in_sizes; (void)n_in; (void)out_size;
  // Input layout: depth-first insertion order of setup_inputs dict.
  // 0:x 1:ip.w 2:ip.b 3:pe
  // enc0: 4:qkv.w 5:qkv.b 6:fc.w 7:fc.b 8:f1.w 9:f1.b 10:f2.w 11:f2.b
  //       12:n1.g 13:n1.b 14:n2.g 15:n2.b      enc1: 16..27 (same pattern)
  // dec : 28:sqkv.w 29:b 30:sfc.w 31:b 32:cqkv.w 33:b 34:cfc.w 35:b
  //       36:f1.w 37:b 38:f2.w 39:b 40:n1.g 41:n1.b 42:n2.g 43:n2.b 44:n3.g 45:n3.b
  // 46:mem.w 47:mem.b (unused) 48:out.w 49:out.b
  auto F = [&](int i) -> const float* { return (const float*)d_in[i]; };
  const float* x = F(0);

  size_t off = 0;
  auto carve = [&](size_t bytes) -> void* {
    size_t o = (off + 255) & ~(size_t)255;
    off = o + bytes;
    return (void*)((char*)d_ws + o);
  };

  // bf16 [N][K] weight mirrors
  struct WD { int idx, K, N; };
  const WD wd[14] = {
      {4, 512, 1536},  {6, 512, 512},  {8, 512, 2048},  {10, 2048, 512},
      {16, 512, 1536}, {18, 512, 512}, {20, 512, 2048}, {22, 2048, 512},
      {28, 512, 1536}, {30, 512, 512}, {32, 512, 1536}, {34, 512, 512},
      {36, 512, 2048}, {38, 2048, 512}};
  unsigned short* wt[14];
  for (int i = 0; i < 14; ++i)
    wt[i] = (unsigned short*)carve((size_t)wd[i].K * wd[i].N * 2);
  for (int i = 0; i < 14; ++i) {
    int total = wd[i].K * wd[i].N;
    k_wconv_t<<<(total + 255) / 256, 256, 0, stream>>>(F(wd[i].idx), wt[i],
                                                       wd[i].K, wd[i].N);
  }

  // batch-chunk sizing: per padded row
  //   h f32 2048B + hbf 1024B + tbig_bf 4096B + ta_bf 1024B + tb f32 2048B
  const size_t per_row = 10240;
  size_t rem = (ws_size > off + 4096) ? (ws_size - off - 4096) : 0;
  long rows_cap = (long)(rem / per_row);
  int Bc = (rows_cap > 63 + 15) ? (int)((rows_cap - 63) / 15) : 1;
  if (Bc < 1) Bc = 1;
  if (Bc > 4096) Bc = 4096;
  int MpadMax = ((Bc * 15 + 63) / 64) * 64;

  float*          h    = (float*)carve((size_t)MpadMax * 512 * 4);
  unsigned short* hbf  = (unsigned short*)carve((size_t)MpadMax * 512 * 2);
  unsigned short* tbig = (unsigned short*)carve((size_t)MpadMax * 2048 * 2);
  unsigned short* ta   = (unsigned short*)carve((size_t)MpadMax * 512 * 2);
  float*          tb   = (float*)carve((size_t)MpadMax * 512 * 4);

  for (int b0 = 0; b0 < 4096; b0 += Bc) {
    const int Bcur = imin(Bc, 4096 - b0);
    const int Mc   = Bcur * 15;
    const int Mpad = ((Mc + 63) / 64) * 64;

    auto gemm = [&](const unsigned short* A, const unsigned short* W,
                    const float* bias, void* C, int N, int K, int mode) {
      dim3 g(Mpad / 64, N / 64);
      k_gemm_bf16<<<g, 256, 0, stream>>>(A, W, bias, C, Mc, N, K, mode);
    };
    auto attn = [&](const unsigned short* qkvb, unsigned short* o) {
      k_attn<<<(Bcur * 8 + 3) / 4, 128, 0, stream>>>(qkvb, o, Bcur);
    };
    auto lnorm = [&](const float* t, const float* g_, const float* b_) {
      k_add_ln<<<(Mc + 7) / 8, 256, 0, stream>>>(h, hbf, t, g_, b_, Mc);
    };

    // input projection + positional encoding (f32 + bf16 mirror)
    k_input_proj<<<Mc, 128, 0, stream>>>(x + (size_t)b0 * 15 * 16, F(1), F(2),
                                         F(3), h, hbf);
    // 2 encoder layers
    for (int L = 0; L < 2; ++L) {
      const int base = (L == 0) ? 4 : 16;
      gemm(hbf, wt[L * 4 + 0], F(base + 1), tbig, 1536, 512, 2);   // qkv (bf16)
      attn(tbig, ta);
      gemm(ta, wt[L * 4 + 1], F(base + 3), tb, 512, 512, 0);       // attn fc
      lnorm(tb, F(base + 8), F(base + 9));                         // norm1
      gemm(hbf, wt[L * 4 + 2], F(base + 5), tbig, 2048, 512, 3);   // ffn1+relu
      gemm(tbig, wt[L * 4 + 3], F(base + 7), tb, 512, 2048, 0);    // ffn2
      lnorm(tb, F(base + 10), F(base + 11));                       // norm2
    }
    // decoder layer (cross-attn is self-attn on x; memory unused)
    gemm(hbf, wt[8], F(29), tbig, 1536, 512, 2);
    attn(tbig, ta);
    gemm(ta, wt[9], F(31), tb, 512, 512, 0);
    lnorm(tb, F(40), F(41));
    gemm(hbf, wt[10], F(33), tbig, 1536, 512, 2);
    attn(tbig, ta);
    gemm(ta, wt[11], F(35), tb, 512, 512, 0);
    lnorm(tb, F(42), F(43));
    gemm(hbf, wt[12], F(37), tbig, 2048, 512, 3);
    gemm(tbig, wt[13], F(39), tb, 512, 2048, 0);
    lnorm(tb, F(44), F(45));

    // mean over seq + output projection
    k_mean_out<<<(Bcur + 7) / 8, 256, 0, stream>>>(h, F(48), F(49),
                                                   (float*)d_out + b0, Bcur);
  }
}
